// TauRoPEAttentionLayer_46557445489097
// MI455X (gfx1250) — compile-verified, matrix operand order checked by reading, not run on hardware
//
#include <hip/hip_runtime.h>
#include <hip/hip_bf16.h>

typedef unsigned short u16;
typedef unsigned int   u32;
typedef unsigned long long u64;
typedef __attribute__((ext_vector_type(16))) __bf16 v16bf;
typedef __attribute__((ext_vector_type(8)))  float  v8f;

struct Frag { u32 r[8]; };

static __device__ __forceinline__ u16 f2bf(float f) {
  u32 u = __builtin_bit_cast(u32, f);
  u32 r = u + 0x7FFFu + ((u >> 16) & 1u);   // round-to-nearest-even
  return (u16)(r >> 16);
}
static __device__ __forceinline__ float bf2f(u16 h) {
  u32 u = ((u32)h) << 16;
  return __builtin_bit_cast(float, u);
}
static __device__ __forceinline__ v8f wmma_bf16(const Frag& a, const Frag& b, v8f c) {
  return __builtin_amdgcn_wmma_f32_16x16x32_bf16(
      false, __builtin_bit_cast(v16bf, a),
      false, __builtin_bit_cast(v16bf, b),
      (short)0, c, false, false);
}
static __device__ __forceinline__ float gelu_exact(float x) {
  return 0.5f * x * (1.0f + erff(x * 0.70710678118654752f));
}

// ---------------------------------------------------------------- TDM (Tensor Data Mover)
#if defined(__HIP_DEVICE_COMPILE__) && __has_builtin(__builtin_amdgcn_tensor_load_to_lds)
#define HAVE_TDM 1
#else
#define HAVE_TDM 0
#endif

static __device__ __forceinline__ u32 lds_off(const void* p) {
  // flat shared-aperture address: low 32 bits are the LDS byte address (ISA 10.2)
  return (u32)(size_t)p;
}
static __device__ __forceinline__ void wait_tensorcnt0() {
#if __has_builtin(__builtin_amdgcn_s_wait_tensorcnt)
  __builtin_amdgcn_s_wait_tensorcnt(0);
#else
  asm volatile("s_wait_tensorcnt 0x0" ::: "memory");
#endif
}

#if HAVE_TDM
typedef __attribute__((ext_vector_type(4))) unsigned int u32x4;
typedef __attribute__((ext_vector_type(8))) int i32x8;
typedef __attribute__((ext_vector_type(4))) int i32x4;

// 2D tile load, bf16 elements, optional LDS row padding.
// pad_code: interval 2^(pi+1) DWORDs, amount (pa+1) DWORDs.
static __device__ __forceinline__ void tdm_load_2d_bf16(
    const void* gtile, u32 lds_byte,
    u32 tensor_d0, u32 tensor_d1, u64 stride0_elems,
    u32 tile_d0, u32 tile_d1, u32 pi, u32 pa)
{
  u64 ga = (u64)(size_t)gtile;
  u32x4 g0;
  g0[0] = 1u;                                            // count=1 user descriptor
  g0[1] = lds_byte;                                      // lds_addr
  g0[2] = (u32)ga;                                       // global_addr[31:0]
  g0[3] = (u32)((ga >> 32) & 0x01FFFFFFu) | 0x80000000u; // global_addr[56:32], type=2
  i32x8 g1;
  g1[0] = (int)((1u << 16) | (1u << 20) | (pi << 22) | (pa << 25)); // data_size=2B, pad_enable
  g1[1] = (int)((tensor_d0 & 0xFFFFu) << 16);            // tensor_dim0[15:0] @ bits63:48
  g1[2] = (int)((tensor_d0 >> 16) | ((tensor_d1 & 0xFFFFu) << 16));
  g1[3] = (int)((tensor_d1 >> 16) | (tile_d0 << 16));    // tile_dim0 @ bits127:112
  g1[4] = (int)(tile_d1 & 0xFFFFu);                      // tile_dim1; tile_dim2=0
  g1[5] = (int)(u32)stride0_elems;                       // tensor_dim0_stride[31:0]
  g1[6] = (int)((u32)(stride0_elems >> 32) & 0xFFFFu);   // stride0[47:32]; stride1=0
  g1[7] = 0;
  i32x4 z = {0, 0, 0, 0};
#if __clang_major__ >= 23
  i32x8 z8 = {0, 0, 0, 0, 0, 0, 0, 0};
  __builtin_amdgcn_tensor_load_to_lds(g0, g1, z, z, z8, 0);
#else
  __builtin_amdgcn_tensor_load_to_lds(g0, g1, z, z, 0);
#endif
}
#endif

// ---------------------------------------------------------------- convert
__global__ void k_f32_to_bf16(const float* __restrict__ in, u16* __restrict__ out, int n) {
  int i = blockIdx.x * blockDim.x + threadIdx.x;
  if (i < n) out[i] = f2bf(in[i]);
}

// ---------------------------------------------------------------- GEMM (bf16 WMMA, f32 acc)
// C[M,N] = A[M,K] * W[K,N] + bias ; epi: 0 = f32 out, 1 = bf16 out, 2 = gelu -> bf16 out
#define G_LDT 34   // LDS row stride (bf16): 32 data + 1-DWORD pad; 17 banks, gcd(17,64)=1

__global__ __launch_bounds__(256) void k_gemm_bf16(
    const u16* __restrict__ A, const u16* __restrict__ W,
    const float* __restrict__ bias,
    float* __restrict__ outF, u16* __restrict__ outB,
    int M, int N, int K, int epi)
{
  __shared__ u16 lA[128 * G_LDT];
  __shared__ u16 lB[128 * G_LDT];

  const int t    = threadIdx.x;
  const int lane = t & 31, wid = t >> 5;
  const int bm   = blockIdx.y * 128, bn = blockIdx.x * 128;
  const int wm   = (wid >> 2) * 64;   // wave rows in tile
  const int wn   = (wid & 3)  * 32;   // wave cols in tile

  v8f acc[4][2];
#pragma unroll
  for (int i = 0; i < 4; ++i)
#pragma unroll
    for (int j = 0; j < 2; ++j) acc[i][j] = {};

  const int arow = t >> 1, acol = (t & 1) * 16;      // fallback A loader
  const int bk2  = (t >> 4) * 2, bns = (t & 15) * 8; // W: 2 k-rows x 8 n each
  const int lm   = lane & 15, kh = lane >> 4;

  for (int k0 = 0; k0 < K; k0 += 32) {
#if HAVE_TDM
    // A tile (128x32, row stride K) via Tensor Data Mover; pad 16-DWORD rows by 1 DWORD
    if (wid == 0)
      tdm_load_2d_bf16(A + (size_t)bm * K + k0, lds_off(&lA[0]),
                       (u32)K, (u32)M, (u64)K, 32u, 128u, /*pi=*/3u, /*pa=*/0u);
#else
    {
      const u16* ag = A + (size_t)(bm + arow) * K + k0 + acol;
      uint4 a0 = *(const uint4*)(ag);
      uint4 a1 = *(const uint4*)(ag + 8);
      u32 aw[8] = {a0.x, a0.y, a0.z, a0.w, a1.x, a1.y, a1.z, a1.w};
#pragma unroll
      for (int i = 0; i < 8; ++i) *(u32*)&lA[arow * G_LDT + acol + 2 * i] = aw[i];
    }
#endif

    // stage W tile transposed [n][k] (k-pairs packed per dword)
    const u16* bg0 = W + (size_t)(k0 + bk2) * N + bn + bns;
    uint4 q0 = *(const uint4*)(bg0);
    uint4 q1 = *(const uint4*)(bg0 + N);
    u32 lo[4] = {q0.x, q0.y, q0.z, q0.w};
    u32 hi[4] = {q1.x, q1.y, q1.z, q1.w};
#pragma unroll
    for (int i = 0; i < 4; ++i) {
      int n0 = bns + 2 * i;
      u32 p0 = (lo[i] & 0xFFFFu) | (hi[i] << 16);
      u32 p1 = (lo[i] >> 16)     | (hi[i] & 0xFFFF0000u);
      *(u32*)&lB[(n0)     * G_LDT + bk2] = p0;
      *(u32*)&lB[(n0 + 1) * G_LDT + bk2] = p1;
    }
    if (k0 + 32 < K)           // prefetch next W tile (global_prefetch_b8)
      __builtin_prefetch(bg0 + (size_t)32 * N, 0, 0);

#if HAVE_TDM
    if (wid == 0) wait_tensorcnt0();
#endif
    __syncthreads();

    Frag af[4], bf[2];
#pragma unroll
    for (int i = 0; i < 4; ++i) {
      const u16* p = &lA[(wm + i * 16 + lm) * G_LDT + kh * 8];
#pragma unroll
      for (int r = 0; r < 4; ++r) {
        af[i].r[r]     = *(const u32*)(p + 2 * r);
        af[i].r[r + 4] = *(const u32*)(p + 16 + 2 * r);
      }
    }
#pragma unroll
    for (int j = 0; j < 2; ++j) {
      const u16* p = &lB[(wn + j * 16 + lm) * G_LDT + kh * 8];
#pragma unroll
      for (int r = 0; r < 4; ++r) {
        bf[j].r[r]     = *(const u32*)(p + 2 * r);
        bf[j].r[r + 4] = *(const u32*)(p + 16 + 2 * r);
      }
    }
#pragma unroll
    for (int i = 0; i < 4; ++i)
#pragma unroll
      for (int j = 0; j < 2; ++j) acc[i][j] = wmma_bf16(af[i], bf[j], acc[i][j]);
    __syncthreads();
  }

  // epilogue: C layout = lane -> N, vgpr r -> M=r (+8 for upper lane half)
  const int lh = lane >> 4;
#pragma unroll
  for (int j = 0; j < 2; ++j) {
    int gn = bn + wn + j * 16 + lm;
    float bv = bias ? bias[gn] : 0.0f;
#pragma unroll
    for (int i = 0; i < 4; ++i) {
      int gmb = bm + wm + i * 16 + lh * 8;
#pragma unroll
      for (int r = 0; r < 8; ++r) {
        float v = acc[i][j][r] + bv;
        size_t o = (size_t)(gmb + r) * N + gn;
        if (epi == 0)      outF[o] = v;
        else if (epi == 1) outB[o] = f2bf(v);
        else               outB[o] = f2bf(gelu_exact(v));
      }
    }
  }
}

// ---------------------------------------------------------------- RoPE (in place, bf16), folds 1/sqrt(d) into Q
__global__ void k_rope(u16* __restrict__ Q, u16* __restrict__ Kk,
                       const float* __restrict__ tau, int total /* B*S*512 */) {
  int idx = blockIdx.x * blockDim.x + threadIdx.x;
  if (idx >= total) return;
  int p = idx & 511;          // pair index within row of 1024
  int row = idx >> 9;         // b*S + s
  int dpair = p & 31;         // pair within head (d/2 = 32)
  float inv = __expf(-((float)(2 * dpair) / 64.0f) * 9.210340371976184f); // 10000^{-2i/64}
  float ang = tau[row] * inv;
  float c = __cosf(ang), s = __sinf(ang);
  size_t off = ((size_t)row << 10) + 2 * p;
  float q0 = bf2f(Q[off]), q1 = bf2f(Q[off + 1]);
  float k0 = bf2f(Kk[off]), k1 = bf2f(Kk[off + 1]);
  const float sc = 0.125f;    // 1/sqrt(64)
  float qe = (q0 * c - q1 * s) * sc, qo = (q1 * c + q0 * s) * sc;
  float ke = (k0 * c - k1 * s),      ko = (k1 * c + k0 * s);
  *(u32*)&Q[off]  = ((u32)f2bf(qo) << 16) | f2bf(qe);
  *(u32*)&Kk[off] = ((u32)f2bf(ko) << 16) | f2bf(ke);
}

// ---------------------------------------------------------------- flash attention (bf16 WMMA)
#define A_LDK 66   // K tile [kj][d=64] stride: 32-DWORD rows + 1-DWORD pad
#define A_LDV 34   // V^T tile [d][kj=32] stride
#define A_LDP 34   // P tile [q=16][kj=32] stride

__global__ __launch_bounds__(128) void k_attn(
    const u16* __restrict__ Q, const u16* __restrict__ K,
    const u16* __restrict__ V, u16* __restrict__ ctx,
    int S /*2048*/, int D /*1024*/)
{
  __shared__ u16 lK[32 * A_LDK];
  __shared__ u16 lV[64 * A_LDV];
  __shared__ u16 lP[4 * 16 * A_LDP];

  const int t = threadIdx.x, lane = t & 31, wid = t >> 5;
  const int b = blockIdx.y >> 4, h = blockIdx.y & 15;
  const int qbase = blockIdx.x * 64 + wid * 16;
  const int lm = lane & 15, kh = lane >> 4, lh = kh;

  // Q fragments (2 d-chunks of 32), gathered straight from global
  Frag qf[2];
  {
    const u16* qg = Q + ((size_t)(b * S + qbase + lm)) * D + h * 64;
#pragma unroll
    for (int fd = 0; fd < 2; ++fd) {
      const u16* p = qg + fd * 32 + kh * 8;
#pragma unroll
      for (int r = 0; r < 4; ++r) {
        qf[fd].r[r]     = *(const u32*)(p + 2 * r);
        qf[fd].r[r + 4] = *(const u32*)(p + 16 + 2 * r);
      }
    }
  }

  v8f cacc[4];
#pragma unroll
  for (int i = 0; i < 4; ++i) cacc[i] = {};
  float Mrow[8], Lrow[8];
#pragma unroll
  for (int r = 0; r < 8; ++r) { Mrow[r] = -1e30f; Lrow[r] = 0.0f; }

  const int kkj = t >> 2, kds = (t & 3) * 16;        // fallback K loader
  const int vkj2 = (t >> 3) * 2, vds = (t & 7) * 8;  // V loader: 2 kj rows x 8 d

  for (int kb = 0; kb < S; kb += 32) {
#if HAVE_TDM
    // K tile (32 x 64, row stride D) via TDM; pad 32-DWORD rows by 1 DWORD
    if (wid == 0)
      tdm_load_2d_bf16(K + ((size_t)(b * S + kb)) * D + h * 64, lds_off(&lK[0]),
                       64u, 32u, (u64)D, 64u, 32u, /*pi=*/4u, /*pa=*/0u);
#else
    {
      const u16* kg = K + ((size_t)(b * S + kb + kkj)) * D + h * 64 + kds;
      uint4 a0 = *(const uint4*)(kg);
      uint4 a1 = *(const uint4*)(kg + 8);
      u32 w[8] = {a0.x, a0.y, a0.z, a0.w, a1.x, a1.y, a1.z, a1.w};
#pragma unroll
      for (int i = 0; i < 8; ++i) *(u32*)&lK[kkj * A_LDK + kds + 2 * i] = w[i];
    }
#endif
    // stage V^T tile [d][kj] (kj-pairs packed)
    {
      const u16* vg0 = V + ((size_t)(b * S + kb + vkj2)) * D + h * 64 + vds;
      const u16* vg1 = vg0 + D;
      uint4 r0 = *(const uint4*)(vg0);
      uint4 r1 = *(const uint4*)(vg1);
      u32 lo[4] = {r0.x, r0.y, r0.z, r0.w};
      u32 hi[4] = {r1.x, r1.y, r1.z, r1.w};
#pragma unroll
      for (int j = 0; j < 8; ++j) {
        u32 a = (lo[j >> 1] >> (16 * (j & 1))) & 0xFFFFu;
        u32 c = (hi[j >> 1] >> (16 * (j & 1))) & 0xFFFFu;
        *(u32*)&lV[(vds + j) * A_LDV + vkj2] = a | (c << 16);
      }
    }
#if HAVE_TDM
    if (wid == 0) wait_tensorcnt0();
#endif
    __syncthreads();

    // logits: two 16x16 key tiles, WMMA over d (2 x k=32)
    v8f s[2];
#pragma unroll
    for (int kt = 0; kt < 2; ++kt) {
      v8f c = {};
#pragma unroll
      for (int fd = 0; fd < 2; ++fd) {
        Frag kf;
        const u16* p = &lK[(kt * 16 + lm) * A_LDK + fd * 32 + kh * 8];
#pragma unroll
        for (int r = 0; r < 4; ++r) {
          kf.r[r]     = *(const u32*)(p + 2 * r);
          kf.r[r + 4] = *(const u32*)(p + 16 + 2 * r);
        }
        c = wmma_bf16(qf[fd], kf, c);
      }
      s[kt] = c;
    }

    // online softmax per row (row = r + lh*8, stats redundant across the 16-lane half)
#pragma unroll
    for (int r = 0; r < 8; ++r) {
      float x = fmaxf(s[0][r], s[1][r]);
      x = fmaxf(x, __shfl_xor(x, 1));
      x = fmaxf(x, __shfl_xor(x, 2));
      x = fmaxf(x, __shfl_xor(x, 4));
      x = fmaxf(x, __shfl_xor(x, 8));
      float mN = fmaxf(Mrow[r], x);
      float alpha = __expf(Mrow[r] - mN);
      Mrow[r] = mN;
      float p0 = __expf(s[0][r] - mN);
      float p1 = __expf(s[1][r] - mN);
      s[0][r] = p0; s[1][r] = p1;
      float sm = p0 + p1;
      sm += __shfl_xor(sm, 1);
      sm += __shfl_xor(sm, 2);
      sm += __shfl_xor(sm, 4);
      sm += __shfl_xor(sm, 8);
      Lrow[r] = Lrow[r] * alpha + sm;
#pragma unroll
      for (int i = 0; i < 4; ++i) cacc[i][r] *= alpha;
    }

    // P (C layout) -> LDS bf16 -> A fragment  (per-wave region; DS is in-order per wave)
    u16* pw = &lP[wid * 16 * A_LDP];
#pragma unroll
    for (int kt = 0; kt < 2; ++kt)
#pragma unroll
      for (int r = 0; r < 8; ++r)
        pw[(r + lh * 8) * A_LDP + kt * 16 + lm] = f2bf(s[kt][r]);

    Frag pf;
    {
      const u16* p = &pw[lm * A_LDP + kh * 8];
#pragma unroll
      for (int r = 0; r < 4; ++r) {
        pf.r[r]     = *(const u32*)(p + 2 * r);
        pf.r[r + 4] = *(const u32*)(p + 16 + 2 * r);
      }
    }
    // ctx += P * V  (4 d-tiles of 16)
#pragma unroll
    for (int dt = 0; dt < 4; ++dt) {
      Frag vf;
      const u16* p = &lV[(dt * 16 + lm) * A_LDV + kh * 8];
#pragma unroll
      for (int r = 0; r < 4; ++r) {
        vf.r[r]     = *(const u32*)(p + 2 * r);
        vf.r[r + 4] = *(const u32*)(p + 16 + 2 * r);
      }
      cacc[dt] = wmma_bf16(pf, vf, cacc[dt]);
    }
    __syncthreads();
  }

  // normalize and write ctx (bf16, merged-head layout [b][s][h*64+d])
#pragma unroll
  for (int dt = 0; dt < 4; ++dt) {
#pragma unroll
    for (int r = 0; r < 8; ++r) {
      int q = qbase + r + lh * 8;
      int d = dt * 16 + lm;
      ctx[((size_t)(b * S + q)) * D + h * 64 + d] = f2bf(cacc[dt][r] / Lrow[r]);
    }
  }
}

// ---------------------------------------------------------------- residual add + LayerNorm
__global__ __launch_bounds__(256) void k_add_ln(
    const float* __restrict__ resid, const float* __restrict__ x,
    const float* __restrict__ g, const float* __restrict__ b,
    float* __restrict__ outF, u16* __restrict__ outB, int D /*1024*/)
{
  __shared__ float red[256];
  int row = blockIdx.x, t = threadIdx.x;
  const float* r0 = resid + (size_t)row * D;
  const float* x0 = x + (size_t)row * D;
  float v[4]; float s = 0.0f;
#pragma unroll
  for (int i = 0; i < 4; ++i) { v[i] = r0[t + i * 256] + x0[t + i * 256]; s += v[i]; }
  red[t] = s; __syncthreads();
  for (int o = 128; o > 0; o >>= 1) { if (t < o) red[t] += red[t + o]; __syncthreads(); }
  float mu = red[0] * (1.0f / 1024.0f);
  __syncthreads();
  float vs = 0.0f;
#pragma unroll
  for (int i = 0; i < 4; ++i) { float d = v[i] - mu; vs += d * d; }
  red[t] = vs; __syncthreads();
  for (int o = 128; o > 0; o >>= 1) { if (t < o) red[t] += red[t + o]; __syncthreads(); }
  float rstd = rsqrtf(red[0] * (1.0f / 1024.0f) + 1e-5f);
#pragma unroll
  for (int i = 0; i < 4; ++i) {
    int idx = t + i * 256;
    float y = (v[i] - mu) * rstd * g[idx] + b[idx];
    outF[(size_t)row * D + idx] = y;
    if (outB) outB[(size_t)row * D + idx] = f2bf(y);
  }
}

// ---------------------------------------------------------------- launch
extern "C" void kernel_launch(void* const* d_in, const int* in_sizes, int n_in,
                              void* d_out, int out_size, void* d_ws, size_t ws_size,
                              hipStream_t stream) {
  const int B = 2, S = 2048, D = 1024, DFF = 4096, BS = B * S;
  const float* src = (const float*)d_in[0];
  const float* tau = (const float*)d_in[1];
  const float* Wq = (const float*)d_in[2];  const float* bq = (const float*)d_in[3];
  const float* Wk = (const float*)d_in[4];  const float* bk = (const float*)d_in[5];
  const float* Wv = (const float*)d_in[6];  const float* bv = (const float*)d_in[7];
  const float* Wo = (const float*)d_in[8];  const float* bo = (const float*)d_in[9];
  const float* W1 = (const float*)d_in[10]; const float* b1 = (const float*)d_in[11];
  const float* W2 = (const float*)d_in[12]; const float* b2 = (const float*)d_in[13];
  const float* g1 = (const float*)d_in[14]; const float* be1 = (const float*)d_in[15];
  const float* g2 = (const float*)d_in[16]; const float* be2 = (const float*)d_in[17];

  char* w = (char*)d_ws;
  const size_t MB = 1u << 20;
  u16* srcB = (u16*)(w + 0 * MB);     // 8 MB
  u16* WqB  = (u16*)(w + 8 * MB);     // 2 MB
  u16* WkB  = (u16*)(w + 10 * MB);
  u16* WvB  = (u16*)(w + 12 * MB);
  u16* WoB  = (u16*)(w + 14 * MB);
  u16* W1B  = (u16*)(w + 16 * MB);    // 8 MB
  u16* W2B  = (u16*)(w + 24 * MB);    // 8 MB
  u16* Qb   = (u16*)(w + 32 * MB);    // 8 MB
  u16* Kb   = (u16*)(w + 40 * MB);
  u16* Vb   = (u16*)(w + 48 * MB);
  u16* ctxB = (u16*)(w + 56 * MB);
  u16* x1B  = (u16*)(w + 64 * MB);
  u16* hB   = (u16*)(w + 72 * MB);    // 32 MB
  float* attnF = (float*)(w + 104 * MB); // 16 MB
  float* x1F   = (float*)(w + 120 * MB); // 16 MB
  float* ffF   = (float*)(w + 136 * MB); // 16 MB

  auto conv = [&](const float* a, u16* o, int n) {
    k_f32_to_bf16<<<(n + 255) / 256, 256, 0, stream>>>(a, o, n);
  };
  conv(src, srcB, BS * D);
  conv(Wq, WqB, D * D); conv(Wk, WkB, D * D);
  conv(Wv, WvB, D * D); conv(Wo, WoB, D * D);
  conv(W1, W1B, D * DFF); conv(W2, W2B, DFF * D);

  dim3 gD(D / 128, BS / 128);     // N=1024 GEMMs
  dim3 gF(DFF / 128, BS / 128);   // N=4096 GEMM
  k_gemm_bf16<<<gD, 256, 0, stream>>>(srcB, WqB, bq, nullptr, Qb, BS, D, D, 1);
  k_gemm_bf16<<<gD, 256, 0, stream>>>(srcB, WkB, bk, nullptr, Kb, BS, D, D, 1);
  k_gemm_bf16<<<gD, 256, 0, stream>>>(srcB, WvB, bv, nullptr, Vb, BS, D, D, 1);

  int npairs = BS * (D / 2);
  k_rope<<<(npairs + 255) / 256, 256, 0, stream>>>(Qb, Kb, tau, npairs);

  dim3 gA(S / 64, B * 16);
  k_attn<<<gA, 128, 0, stream>>>(Qb, Kb, Vb, ctxB, S, D);

  k_gemm_bf16<<<gD, 256, 0, stream>>>(ctxB, WoB, bo, attnF, nullptr, BS, D, D, 0);
  k_add_ln<<<BS, 256, 0, stream>>>(src, attnF, g1, be1, x1F, x1B, D);

  k_gemm_bf16<<<gF, 256, 0, stream>>>(x1B, W1B, b1, nullptr, hB, BS, DFF, D, 2); // gelu fused
  k_gemm_bf16<<<gD, 256, 0, stream>>>(hB, W2B, b2, ffF, nullptr, BS, D, DFF, 0);
  k_add_ln<<<BS, 256, 0, stream>>>(x1F, ffF, g2, be2, (float*)d_out, nullptr, D);
}
